// SelfAttention_29850022707361
// MI455X (gfx1250) — matrix operand order repaired and checked by reading, not compiled, not run on hardware
//
#include <hip/hip_runtime.h>
#include <hip/hip_bf16.h>

#define EMBED   1024
#define HEADS   16
#define HDIM    64
#define NBATCH  4
#define SEQ     2048

typedef __attribute__((ext_vector_type(16))) __bf16          v16bf;
typedef __attribute__((ext_vector_type(8)))  float           v8f;
typedef __attribute__((ext_vector_type(8)))  unsigned short  us8;
typedef __attribute__((ext_vector_type(16))) unsigned short  us16;
typedef __attribute__((ext_vector_type(4)))  unsigned int    u32x4;
typedef __attribute__((ext_vector_type(4)))  float           f32x4;

// ---- f32 -> bf16 conversion (native fptrunc, RTNE) ------------------------

static __device__ __forceinline__ unsigned short f2bf1(float a) {
  __bf16 h = (__bf16)a;
  return __builtin_bit_cast(unsigned short, h);
}
static __device__ __forceinline__ unsigned f2bf_pk(float a, float b) {
  return (unsigned)f2bf1(a) | ((unsigned)f2bf1(b) << 16);
}

static __device__ __forceinline__ v8f vzero8() {
  v8f z;
#pragma unroll
  for (int i = 0; i < 8; ++i) z[i] = 0.0f;
  return z;
}

static __device__ __forceinline__ v8f wmma_bf16(v16bf a, v16bf b, v8f c) {
  // D = A(16x32 bf16) x B(32x16 bf16) + C(16x16 f32)
  return __builtin_amdgcn_wmma_f32_16x16x32_bf16(false, a, false, b,
                                                 (short)0, c, false, false);
}

// A fragment (16x32, MxK): lane-lo row l holds K 0..15; lane-hi K 16..31 (contiguous)
static __device__ __forceinline__ v16bf load_fragA_bf16(const unsigned short* base,
                                                        int row0, int col0, int ld) {
  int lane = threadIdx.x & 31;
  int lo = lane & 15, hi = lane >> 4;
  const unsigned short* p = base + (long)(row0 + lo) * ld + col0 + hi * 16;
  union { v16bf v; us8 h[2]; } u;
  u.h[0] = *(const us8*)p;
  u.h[1] = *(const us8*)(p + 8);
  return u.v;
}

// B fragment (32x16, KxN): lane-lo col l holds K 0..15; lane-hi K 16..31 (contiguous)
static __device__ __forceinline__ v16bf load_fragB_bf16(const unsigned short* base,
                                                        int n0, int k0, int ld) {
  int lane = threadIdx.x & 31;
  int lo = lane & 15, hi = lane >> 4;
  const unsigned short* p = base + (long)(n0 + lo) * ld + k0 + hi * 16;
  us16 a = *(const us16*)p;
  return __builtin_bit_cast(v16bf, a);
}

// ---- CDNA5 async global->LDS (ASYNCcnt) -----------------------------------

static __device__ __forceinline__ unsigned lds_off_of(const unsigned short* p) {
  return (unsigned)(unsigned long long)
         (const __attribute__((address_space(3))) unsigned short*)p;
}
static __device__ __forceinline__ void async_cp16(const unsigned short* lds_dst,
                                                  const unsigned short* gsrc) {
  unsigned loff = lds_off_of(lds_dst);
  unsigned long long ga = (unsigned long long)gsrc;
  asm volatile("global_load_async_to_lds_b128 %0, %1, off"
               :: "v"(loff), "v"(ga) : "memory");
}

// ---- kernel 0: bulk f32 -> bf16 conversion (8 elems/thread, b128 stores) --
__global__ void __launch_bounds__(256) cvt8_kernel(const float* __restrict__ src,
                                                   unsigned short* __restrict__ dst,
                                                   int n8) {
  int i = blockIdx.x * blockDim.x + threadIdx.x;
  if (i >= n8) return;
  const float* p = src + (long)i * 8;
  f32x4 a = *(const f32x4*)p;
  f32x4 b = *(const f32x4*)(p + 4);
  u32x4 o;
  o[0] = f2bf_pk(a[0], a[1]);
  o[1] = f2bf_pk(a[2], a[3]);
  o[2] = f2bf_pk(b[0], b[1]);
  o[3] = f2bf_pk(b[2], b[3]);
  *(u32x4*)(dst + (long)i * 8) = o;
}

// ---- kernel 1: per-head QKV projection (x @ W^T), all-bf16 ----------------
// Qp,Kp: [N,H,S,D]; Vt: [N,H,D,S]. Q pre-scaled by 1/sqrt(EMBED) = 1/32.
__global__ void __launch_bounds__(128) proj_kernel(
    const unsigned short* __restrict__ xq, const unsigned short* __restrict__ xk,
    const unsigned short* __restrict__ xv,
    const unsigned short* __restrict__ Wq, const unsigned short* __restrict__ Wk,
    const unsigned short* __restrict__ Wv,
    unsigned short* __restrict__ Qp, unsigned short* __restrict__ Kp,
    unsigned short* __restrict__ Vt) {
  int wv  = blockIdx.x * (blockDim.x >> 5) + (threadIdx.x >> 5);
  int s16 = wv & (SEQ / 16 - 1);
  int h   = (wv >> 7) & (HEADS - 1);
  int n   = wv >> 11;
  int lane = threadIdx.x & 31;
  int lo = lane & 15, hi = lane >> 4;

  long inoff = ((long)n * SEQ + (long)s16 * 16) * EMBED + (long)h * HDIM;
  v16bf aq0 = load_fragA_bf16(xq + inoff, 0, 0,  EMBED);
  v16bf aq1 = load_fragA_bf16(xq + inoff, 0, 32, EMBED);
  v16bf ak0 = load_fragA_bf16(xk + inoff, 0, 0,  EMBED);
  v16bf ak1 = load_fragA_bf16(xk + inoff, 0, 32, EMBED);
  v16bf av0 = load_fragA_bf16(xv + inoff, 0, 0,  EMBED);
  v16bf av1 = load_fragA_bf16(xv + inoff, 0, 32, EMBED);

  long hdbase = (((long)n * HEADS + h) * SEQ + (long)s16 * 16) * HDIM;
  long vtbase = ((long)n * HEADS + h) * (long)HDIM * SEQ;

#pragma unroll
  for (int dt = 0; dt < 4; ++dt) {
    v16bf bq0 = load_fragB_bf16(Wq, dt * 16, 0,  HDIM);
    v16bf bk0 = load_fragB_bf16(Wk, dt * 16, 0,  HDIM);
    v16bf bv0 = load_fragB_bf16(Wv, dt * 16, 0,  HDIM);
    v16bf bq1 = load_fragB_bf16(Wq, dt * 16, 32, HDIM);
    v16bf bk1 = load_fragB_bf16(Wk, dt * 16, 32, HDIM);
    v16bf bv1 = load_fragB_bf16(Wv, dt * 16, 32, HDIM);
    // interleave 3 independent accumulator chains (hides WMMA hazards)
    v8f cq = vzero8(), ck = vzero8(), cv = vzero8();
    cq = wmma_bf16(aq0, bq0, cq);
    ck = wmma_bf16(ak0, bk0, ck);
    cv = wmma_bf16(av0, bv0, cv);
    cq = wmma_bf16(aq1, bq1, cq);
    ck = wmma_bf16(ak1, bk1, ck);
    cv = wmma_bf16(av1, bv1, cv);
#pragma unroll
    for (int j = 0; j < 8; ++j) {
      int row = j + 8 * hi;                     // C layout: M = j + 8*hi, N = lo
      int col = dt * 16 + lo;
      Qp[hdbase + (long)row * HDIM + col] = f2bf1(cq[j] * 0.03125f);
      Kp[hdbase + (long)row * HDIM + col] = f2bf1(ck[j]);
      Vt[vtbase + (long)col * SEQ + s16 * 16 + row] = f2bf1(cv[j]);
    }
  }
}

// ---- kernel 2: fused flash attention with double-buffered async K/V -------
__global__ void __launch_bounds__(128) attn_kernel(
    const unsigned short* __restrict__ Qp, const unsigned short* __restrict__ Kp,
    const unsigned short* __restrict__ Vt, const int* __restrict__ masks,
    unsigned short* __restrict__ AttnO) {
  __shared__ __align__(32) unsigned short KtileL[2][64 * 64];  // 2 x 8 KB
  __shared__ __align__(32) unsigned short VtileL[2][64 * 64];  // 2 x 8 KB
  __shared__ __align__(32) unsigned short Plds[4][16 * 64];    // per-wave P

  int t    = threadIdx.x;
  int w    = t >> 5;
  int lane = t & 31;
  int lo = lane & 15, hi = lane >> 4;

  int blk = blockIdx.x;                 // N*H*(S/64) = 2048 blocks
  int qt  = blk & (SEQ / 64 - 1);
  int h   = (blk >> 5) & (HEADS - 1);
  int n   = blk >> 9;
  int q0  = qt * 64 + w * 16;

  const unsigned short* Qh = Qp + ((long)n * HEADS + h) * SEQ * HDIM;
  const unsigned short* Kh = Kp + ((long)n * HEADS + h) * SEQ * HDIM;
  const unsigned short* Vh = Vt + ((long)n * HEADS + h) * (long)HDIM * SEQ;
  const int* mk = masks + (long)n * SEQ;

  // cooperative async stage of one K tile (contiguous 8 KB) + one V tile
  auto issue_tile = [&](unsigned short* kl, unsigned short* vl, int kb) {
    const unsigned short* kg = Kh + (long)kb * HDIM;
#pragma unroll
    for (int i = 0; i < 4; ++i)
      async_cp16(kl + (i * 128 + t) * 8, kg + (i * 128 + t) * 8);
    int row = t >> 1, half = t & 1;
    const unsigned short* vg = Vh + (long)row * SEQ + kb + half * 32;
    unsigned short* vd = vl + row * 64 + half * 32;
#pragma unroll
    for (int i = 0; i < 4; ++i)
      async_cp16(vd + i * 8, vg + i * 8);
  };

  v16bf aq0 = load_fragA_bf16(Qh, q0, 0,  HDIM);
  v16bf aq1 = load_fragA_bf16(Qh, q0, 32, HDIM);

  v8f o[4];
#pragma unroll
  for (int dt = 0; dt < 4; ++dt) o[dt] = vzero8();
  float mrun[8], lrun[8];
#pragma unroll
  for (int j = 0; j < 8; ++j) { mrun[j] = -1e30f; lrun[j] = 0.0f; }

  issue_tile(KtileL[0], VtileL[0], 0);     // prologue: tile 0 in flight
  int buf = 0;

  for (int it = 0; it < SEQ / 64; ++it) {
    int kb = it * 64;
    __syncthreads();  // all waves done reading buf^1 (previous iteration)
    issue_tile(KtileL[buf ^ 1], VtileL[buf ^ 1], (kb + 64) & (SEQ - 1));
    asm volatile("s_wait_asynccnt 0x8" ::: "memory");  // tile `it` landed
    __syncthreads();  // every wave's portion of tile `it` visible
    const unsigned short* Kl = &KtileL[buf][0];
    const unsigned short* Vl = &VtileL[buf][0];

    // ---- energy tile E = (Q/32) x K^T : 16 x 64 (4 independent chains) ----
    v8f e[4];
#pragma unroll
    for (int nt = 0; nt < 4; ++nt) e[nt] = vzero8();
#pragma unroll
    for (int kc = 0; kc < 2; ++kc) {
      v16bf b0 = load_fragB_bf16(Kl, 0,  kc * 32, 64);
      v16bf b1 = load_fragB_bf16(Kl, 16, kc * 32, 64);
      v16bf b2 = load_fragB_bf16(Kl, 32, kc * 32, 64);
      v16bf b3 = load_fragB_bf16(Kl, 48, kc * 32, 64);
      v16bf a = kc ? aq1 : aq0;
      e[0] = wmma_bf16(a, b0, e[0]);
      e[1] = wmma_bf16(a, b1, e[1]);
      e[2] = wmma_bf16(a, b2, e[2]);
      e[3] = wmma_bf16(a, b3, e[3]);
    }
    float s[4][8];
#pragma unroll
    for (int nt = 0; nt < 4; ++nt) {
      int mv = mk[kb + nt * 16 + lo];   // mask broadcast down the column
#pragma unroll
      for (int j = 0; j < 8; ++j)
        s[nt][j] = mv ? e[nt][j] : -3.125e18f;   // (-1e20)/32
    }
    // ---- row max across 64 cols ----
    float tmax[8];
#pragma unroll
    for (int j = 0; j < 8; ++j)
      tmax[j] = fmaxf(fmaxf(s[0][j], s[1][j]), fmaxf(s[2][j], s[3][j]));
#pragma unroll
    for (int m = 1; m <= 8; m <<= 1)
#pragma unroll
      for (int j = 0; j < 8; ++j)
        tmax[j] = fmaxf(tmax[j], __shfl_xor(tmax[j], m, 32));

    float mnew[8], resc[8], lsum[8];
#pragma unroll
    for (int j = 0; j < 8; ++j) {
      mnew[j] = fmaxf(mrun[j], tmax[j]);
      resc[j] = __expf(mrun[j] - mnew[j]);
      lsum[j] = 0.0f;
    }
    // ---- P = exp(s - mnew) -> per-wave LDS tile (C->A relayout) ----
#pragma unroll
    for (int nt = 0; nt < 4; ++nt)
#pragma unroll
      for (int j = 0; j < 8; ++j) {
        float p = __expf(s[nt][j] - mnew[j]);
        lsum[j] += p;
        Plds[w][(j + 8 * hi) * 64 + nt * 16 + lo] = f2bf1(p);
      }
#pragma unroll
    for (int m = 1; m <= 8; m <<= 1)
#pragma unroll
      for (int j = 0; j < 8; ++j)
        lsum[j] += __shfl_xor(lsum[j], m, 32);
#pragma unroll
    for (int j = 0; j < 8; ++j) {
      lrun[j] = lrun[j] * resc[j] + lsum[j];
      mrun[j] = mnew[j];
    }
#pragma unroll
    for (int dt = 0; dt < 4; ++dt)
#pragma unroll
      for (int j = 0; j < 8; ++j) o[dt][j] *= resc[j];

    // ---- O += P x V (4 independent chains per kc) ----
    v16bf ap0 = load_fragA_bf16(&Plds[w][0], 0, 0,  64);
    v16bf ap1 = load_fragA_bf16(&Plds[w][0], 0, 32, 64);
#pragma unroll
    for (int kc = 0; kc < 2; ++kc) {
      v16bf b0 = load_fragB_bf16(Vl, 0,  kc * 32, 64);
      v16bf b1 = load_fragB_bf16(Vl, 16, kc * 32, 64);
      v16bf b2 = load_fragB_bf16(Vl, 32, kc * 32, 64);
      v16bf b3 = load_fragB_bf16(Vl, 48, kc * 32, 64);
      v16bf a = kc ? ap1 : ap0;
      o[0] = wmma_bf16(a, b0, o[0]);
      o[1] = wmma_bf16(a, b1, o[1]);
      o[2] = wmma_bf16(a, b2, o[2]);
      o[3] = wmma_bf16(a, b3, o[3]);
    }
    buf ^= 1;
  }

  // ---- epilogue: O *= 1/l, write bf16 [N,S,E] ----
  float inv[8];
#pragma unroll
  for (int j = 0; j < 8; ++j) inv[j] = __builtin_amdgcn_rcpf(lrun[j]);
  long obase = ((long)n * SEQ + q0) * EMBED + (long)h * HDIM;
#pragma unroll
  for (int dt = 0; dt < 4; ++dt)
#pragma unroll
    for (int j = 0; j < 8; ++j)
      AttnO[obase + (long)(j + 8 * hi) * EMBED + dt * 16 + lo] =
          f2bf1(o[dt][j] * inv[j]);
}

// ---- kernel 3: output projection out = A @ Wo^T + bo ----------------------
__global__ void __launch_bounds__(128) outproj_kernel(
    const unsigned short* __restrict__ AttnO, const unsigned short* __restrict__ Wob,
    const float* __restrict__ bo, float* __restrict__ out) {
  int wv = blockIdx.x * (blockDim.x >> 5) + (threadIdx.x >> 5);
  int et = wv & 15;            // 16 col tiles of 64
  int r16 = wv >> 4;           // 512 row tiles of 16
  int r0 = r16 * 16, e0 = et * 64;
  int lane = threadIdx.x & 31;
  int lo = lane & 15, hi = lane >> 4;

  v8f acc[4];
#pragma unroll
  for (int nt = 0; nt < 4; ++nt) acc[nt] = vzero8();

  for (int kc = 0; kc < EMBED; kc += 32) {
    v16bf a = load_fragA_bf16(AttnO, r0, kc, EMBED);
#pragma unroll
    for (int nt = 0; nt < 4; ++nt)
      acc[nt] = wmma_bf16(a, load_fragB_bf16(Wob, e0 + nt * 16, kc, EMBED), acc[nt]);
  }
#pragma unroll
  for (int nt = 0; nt < 4; ++nt) {
    float b = bo[e0 + nt * 16 + lo];
#pragma unroll
    for (int j = 0; j < 8; ++j)
      out[((long)r0 + j + 8 * hi) * EMBED + e0 + nt * 16 + lo] = acc[nt][j] + b;
  }
}

// ---- launcher -------------------------------------------------------------
extern "C" void kernel_launch(void* const* d_in, const int* in_sizes, int n_in,
                              void* d_out, int out_size, void* d_ws, size_t ws_size,
                              hipStream_t stream) {
  const float* q     = (const float*)d_in[0];
  const float* k     = (const float*)d_in[1];
  const float* v     = (const float*)d_in[2];
  const int*   masks = (const int*)  d_in[3];
  const float* Wq    = (const float*)d_in[4];
  const float* Wk    = (const float*)d_in[5];
  const float* Wv    = (const float*)d_in[6];
  const float* Wo    = (const float*)d_in[7];
  const float* bo    = (const float*)d_in[8];
  float* out = (float*)d_out;

  const size_t SZ = (size_t)NBATCH * HEADS * SEQ * HDIM;  // 8M elements
  const size_t WSZ = (size_t)HDIM * HDIM;                  // 4K elements
  unsigned short* ws    = (unsigned short*)d_ws;
  unsigned short* Qp    = ws;            // 16 MB
  unsigned short* Kp    = Qp + SZ;       // 16 MB
  unsigned short* Vt    = Kp + SZ;       // 16 MB
  unsigned short* AttnO = Vt + SZ;       // 16 MB
  unsigned short* Xqb   = AttnO + SZ;    // 16 MB
  unsigned short* Xkb   = Xqb + SZ;      // 16 MB
  unsigned short* Xvb   = Xkb + SZ;      // 16 MB
  unsigned short* Wob   = Xvb + SZ;      //  2 MB
  unsigned short* Wqb   = Wob + (size_t)EMBED * EMBED;  // 8 KB
  unsigned short* Wkb   = Wqb + WSZ;
  unsigned short* Wvb   = Wkb + WSZ;     // total ~114 MB

  cvt8_kernel<<<(int)(SZ / 8 + 255) / 256, 256, 0, stream>>>(q, Xqb, (int)(SZ / 8));
  cvt8_kernel<<<(int)(SZ / 8 + 255) / 256, 256, 0, stream>>>(k, Xkb, (int)(SZ / 8));
  cvt8_kernel<<<(int)(SZ / 8 + 255) / 256, 256, 0, stream>>>(v, Xvb, (int)(SZ / 8));
  cvt8_kernel<<<(EMBED * EMBED / 8 + 255) / 256, 256, 0, stream>>>(Wo, Wob, EMBED * EMBED / 8);
  cvt8_kernel<<<(int)(WSZ / 8 + 255) / 256, 256, 0, stream>>>(Wq, Wqb, (int)(WSZ / 8));
  cvt8_kernel<<<(int)(WSZ / 8 + 255) / 256, 256, 0, stream>>>(Wk, Wkb, (int)(WSZ / 8));
  cvt8_kernel<<<(int)(WSZ / 8 + 255) / 256, 256, 0, stream>>>(Wv, Wvb, (int)(WSZ / 8));

  proj_kernel<<<2048, 128, 0, stream>>>(Xqb, Xkb, Xvb, Wqb, Wkb, Wvb, Qp, Kp, Vt);
  attn_kernel<<<2048, 128, 0, stream>>>(Qp, Kp, Vt, masks, AttnO);
  outproj_kernel<<<2048, 128, 0, stream>>>(AttnO, Wob, bo, out);
}
